// EvoformerGenomeModel_29240137351814
// MI455X (gfx1250) — matrix operand order repaired and checked by reading, not compiled, not run on hardware
//
#include <hip/hip_runtime.h>
#include <math.h>

#define L_ 384
#define D_ 256
#define P_ 128
#define H_ 8
#define HP_ 4
#define F_ 1024

typedef __attribute__((ext_vector_type(16))) __bf16 v16bf;
typedef __attribute__((ext_vector_type(8)))  __bf16 v8bf;
typedef __attribute__((ext_vector_type(8)))  float  v8f;

// ---------------- device helpers ----------------
__device__ __forceinline__ unsigned short f2bf(float f) {
  union { float f; unsigned int u; } v; v.f = f;
  unsigned int u = v.u;
  u += 0x7FFFu + ((u >> 16) & 1u);   // round-to-nearest-even
  return (unsigned short)(u >> 16);
}
__device__ __forceinline__ float geluf(float x) {
  return 0.5f * x * (1.0f + erff(x * 0.7071067811865476f));
}
__device__ __forceinline__ float sigm(float x) { return 1.0f / (1.0f + expf(-x)); }

__device__ __forceinline__ v16bf zero_frag() {
  v16bf r;
#pragma unroll
  for (int t = 0; t < 16; ++t) r[t] = (__bf16)0.0f;
  return r;
}
// A fragment, unit-stride K: two 16B chunks at +0 and +8 elements
__device__ __forceinline__ v16bf load_fragA_c(const __bf16* p) {
  v8bf a0 = *(const v8bf*)(p);
  v8bf a1 = *(const v8bf*)(p + 8);
  v16bf r;
#pragma unroll
  for (int t = 0; t < 8; ++t) { r[t] = a0[t]; r[8 + t] = a1[t]; }
  return r;
}
// B fragment, unit-stride K: 16 contiguous elements
__device__ __forceinline__ v16bf load_fragB_c(const __bf16* p) {
  v8bf b0 = *(const v8bf*)(p);
  v8bf b1 = *(const v8bf*)(p + 8);
  v16bf r;
#pragma unroll
  for (int t = 0; t < 8; ++t) { r[t] = b0[t]; r[8 + t] = b1[t]; }
  return r;
}
__device__ __forceinline__ v16bf load_fragA_g(const __bf16* p, int ka, int strideK) {
  v16bf r;
#pragma unroll
  for (int t = 0; t < 8; ++t) r[t] = p[(long long)(ka + t) * strideK];
#pragma unroll
  for (int t = 0; t < 8; ++t) r[8 + t] = p[(long long)(ka + 8 + t) * strideK];
  return r;
}
__device__ __forceinline__ v16bf load_fragB_g(const __bf16* p, int kb, int strideK) {
  v16bf r;
#pragma unroll
  for (int t = 0; t < 16; ++t) r[t] = p[(long long)(kb + t) * strideK];
  return r;
}

// ---------------- generic batched WMMA GEMM ----------------
// C[z][m][n] = epilogue( alpha * ( sum_k A[z][m][k] * B[z][n][k] + bias[n] ) )
// A addressed: A + zo*aOut + zi*aIn + m*aRow + k*aK      (bf16)
// B addressed: B + zo*bOut + zi*bIn + n*bRow + k*bK      (bf16, N x K view)
// C addressed: C + zo*cOut + zi*cIn + m*cRow + n         (f32 or bf16)
// Each wave computes a 16x32 tile (one A fragment feeds two WMMAs).
// AC/BC: compile-time "K is unit-stride" -> 128-bit vector fragment loads.
template <int AC, int BC>
__global__ __launch_bounds__(32)
void gemm_wmma_kernel(const __bf16* __restrict__ A, const __bf16* __restrict__ B,
                      const float* __restrict__ bias, void* __restrict__ C, int cbf,
                      int M, int N, int K,
                      long long aOut, long long aIn, int aRow, int aK,
                      long long bOut, long long bIn, int bRow, int bK,
                      long long cOut, long long cIn, int cRow,
                      int batchInner, float alpha, const float* __restrict__ gate,
                      int accum, int act)
{
  const int lane = threadIdx.x;
  const int tn = blockIdx.x, tm = blockIdx.y, z = blockIdx.z;
  const int zo = z / batchInner, zi = z - zo * batchInner;
  const __bf16* Ab = A + (long long)zo * aOut + (long long)zi * aIn;
  const __bf16* Bb = B + (long long)zo * bOut + (long long)zi * bIn;
  const int half = lane >> 4;
  const int lid  = lane & 15;
  const int m = tm * 16 + lid;
  const int nc0 = tn * 32 + lid;
  const int nc1 = nc0 + 16;

  // contiguous-path pointers (pre-offset by half-wave K sub-block)
  const __bf16* ApC  = Ab + (long long)m * aRow + half * 16;
  const __bf16* BpC0 = Bb + (long long)nc0 * bRow + half * 16;
  const __bf16* BpC1 = Bb + (long long)nc1 * bRow + half * 16;
  // generic-path row pointers
  const __bf16* ApG  = Ab + (long long)m * aRow;
  const __bf16* BpG0 = Bb + (long long)nc0 * bRow;
  const __bf16* BpG1 = Bb + (long long)nc1 * bRow;

  v8f acc0 = {0.f, 0.f, 0.f, 0.f, 0.f, 0.f, 0.f, 0.f};
  v8f acc1 = {0.f, 0.f, 0.f, 0.f, 0.f, 0.f, 0.f, 0.f};

  // wave-uniform full-tile test -> guard-free hot loop
  const bool fullTile = (tm * 16 + 16 <= M) && (tn * 32 + 32 <= N);
  if (fullTile) {
    for (int k0 = 0; k0 < K; k0 += 32) {
      v16bf av  = AC ? load_fragA_c(ApC)  : load_fragA_g(ApG, k0 + half * 16, aK);
      v16bf bv0 = BC ? load_fragB_c(BpC0) : load_fragB_g(BpG0, k0 + half * 16, bK);
      v16bf bv1 = BC ? load_fragB_c(BpC1) : load_fragB_g(BpG1, k0 + half * 16, bK);
      if (AC) ApC += 32;
      if (BC) { BpC0 += 32; BpC1 += 32; }
      acc0 = __builtin_amdgcn_wmma_f32_16x16x32_bf16(false, av, false, bv0,
                                                     (short)0, acc0, false, false);
      acc1 = __builtin_amdgcn_wmma_f32_16x16x32_bf16(false, av, false, bv1,
                                                     (short)0, acc1, false, false);
    }
  } else {
    const bool mok = (m < M);
    const bool n0ok = (nc0 < N);
    const bool n1ok = (nc1 < N);
    for (int k0 = 0; k0 < K; k0 += 32) {
      v16bf av, bv0, bv1;
      if (mok) {
        av = AC ? load_fragA_c(ApC) : load_fragA_g(ApG, k0 + half * 16, aK);
        if (AC) ApC += 32;
      } else av = zero_frag();
      if (n0ok) {
        bv0 = BC ? load_fragB_c(BpC0) : load_fragB_g(BpG0, k0 + half * 16, bK);
        if (BC) BpC0 += 32;
      } else bv0 = zero_frag();
      if (n1ok) {
        bv1 = BC ? load_fragB_c(BpC1) : load_fragB_g(BpG1, k0 + half * 16, bK);
        if (BC) BpC1 += 32;
      } else bv1 = zero_frag();
      acc0 = __builtin_amdgcn_wmma_f32_16x16x32_bf16(false, av, false, bv0,
                                                     (short)0, acc0, false, false);
      acc1 = __builtin_amdgcn_wmma_f32_16x16x32_bf16(false, av, false, bv1,
                                                     (short)0, acc1, false, false);
    }
  }

  float sc = alpha;
  if (gate) sc *= 1.0f / (1.0f + expf(-gate[0]));
  const long long cBase = (long long)zo * cOut + (long long)zi * cIn;
#pragma unroll
  for (int tt = 0; tt < 2; ++tt) {
    const int n = tt ? nc1 : nc0;
    if (n >= N) continue;
    const float bval = bias ? bias[n] : 0.0f;
    const v8f& acc = tt ? acc1 : acc0;
#pragma unroll
    for (int r = 0; r < 8; ++r) {
      const int mi = tm * 16 + r + half * 8;
      if (mi >= M) continue;
      float v = sc * (acc[r] + bval);
      if (act == 1) v = geluf(v);
      const long long idx = cBase + (long long)mi * cRow + n;
      if (cbf) ((unsigned short*)C)[idx] = f2bf(v);
      else {
        float* Cf = (float*)C;
        if (accum) Cf[idx] += v; else Cf[idx] = v;
      }
    }
  }
}

// ---------------- LayerNorm over last dim ----------------
__global__ __launch_bounds__(128)
void ln_kernel(const float* __restrict__ x, const float* __restrict__ g,
               const float* __restrict__ b, float* __restrict__ outF,
               unsigned short* __restrict__ outB, int dim, int Ltrans)
{
  const long long r = blockIdx.x;
  const int tid = threadIdx.x;
  const float* xr = x + r * dim;
  __shared__ float red[128];
  float s = 0.f;
  for (int i = tid; i < dim; i += 128) s += xr[i];
  red[tid] = s; __syncthreads();
  for (int o = 64; o > 0; o >>= 1) { if (tid < o) red[tid] += red[tid + o]; __syncthreads(); }
  const float mu = red[0] / dim;
  __syncthreads();
  float vv = 0.f;
  for (int i = tid; i < dim; i += 128) { float d = xr[i] - mu; vv += d * d; }
  red[tid] = vv; __syncthreads();
  for (int o = 64; o > 0; o >>= 1) { if (tid < o) red[tid] += red[tid + o]; __syncthreads(); }
  const float inv = rsqrtf(red[0] / dim + 1e-5f);
  long long wr = r;
  if (Ltrans) { long long i = r / Ltrans, j = r - i * Ltrans; wr = j * Ltrans + i; }
  for (int i = tid; i < dim; i += 128) {
    const float y = (xr[i] - mu) * inv * g[i] + b[i];
    if (outF) outF[wr * dim + i] = y;
    if (outB) outB[wr * dim + i] = f2bf(y);
  }
}

// LayerNorm over channel-major input [P][LL] -> bf16 (LL,P)
__global__ __launch_bounds__(128)
void ln_ch_kernel(const float* __restrict__ Xc, const float* __restrict__ g,
                  const float* __restrict__ b, unsigned short* __restrict__ outB,
                  long long LLs)
{
  const long long s = blockIdx.x;
  const int d = threadIdx.x;
  const float x = Xc[(long long)d * LLs + s];
  __shared__ float red[128];
  red[d] = x; __syncthreads();
  for (int o = 64; o > 0; o >>= 1) { if (d < o) red[d] += red[d + o]; __syncthreads(); }
  const float mu = red[0] / 128.0f; __syncthreads();
  const float df = x - mu;
  red[d] = df * df; __syncthreads();
  for (int o = 64; o > 0; o >>= 1) { if (d < o) red[d] += red[d + o]; __syncthreads(); }
  const float inv = rsqrtf(red[0] / 128.0f + 1e-5f);
  outB[s * 128 + d] = f2bf(df * inv * g[d] + b[d]);
}

// ---------------- row softmax (optional seq-attention pair bias) ----------------
__global__ __launch_bounds__(128)
void softmax_kernel(float* __restrict__ x, int n, const float* __restrict__ bias,
                    int Lb, int Hb)
{
  const long long r = blockIdx.x;
  const int tid = threadIdx.x;
  float* xr = x + r * n;
  const float* brow = nullptr;
  if (bias) { long long h = r / Lb, l = r - h * Lb; brow = bias + l * Lb * Hb + h; }
  __shared__ float red[128];
  float mx = -3.0e38f;
  for (int i = tid; i < n; i += 128) {
    float v = xr[i] + (brow ? brow[(long long)i * Hb] : 0.0f);
    xr[i] = v;
    mx = fmaxf(mx, v);
  }
  red[tid] = mx; __syncthreads();
  for (int o = 64; o > 0; o >>= 1) { if (tid < o) red[tid] = fmaxf(red[tid], red[tid + o]); __syncthreads(); }
  const float M = red[0]; __syncthreads();
  float s = 0.f;
  for (int i = tid; i < n; i += 128) { float e = expf(xr[i] - M); xr[i] = e; s += e; }
  red[tid] = s; __syncthreads();
  for (int o = 64; o > 0; o >>= 1) { if (tid < o) red[tid] += red[tid + o]; __syncthreads(); }
  const float inv = 1.0f / red[0];
  for (int i = tid; i < n; i += 128) xr[i] *= inv;
}

// ---------------- elementwise kernels ----------------
__global__ void cvt_kernel(const float* __restrict__ x, unsigned short* __restrict__ y, long long n) {
  for (long long i = blockIdx.x * 256LL + threadIdx.x; i < n; i += (long long)gridDim.x * 256)
    y[i] = f2bf(x[i]);
}
__global__ void geglu_kernel(const float* __restrict__ h, unsigned short* __restrict__ y,
                             int rows, int Fh) {
  const long long n = (long long)rows * Fh;
  for (long long i = blockIdx.x * 256LL + threadIdx.x; i < n; i += (long long)gridDim.x * 256) {
    const long long r = i / Fh; const int c = (int)(i - r * Fh);
    const float xv = h[r * 2 * Fh + c];
    const float gv = h[r * 2 * Fh + Fh + c];
    y[i] = f2bf(xv * geluf(gv));
  }
}
// value = a*sigmoid(g), packed channel-major [P][L][L] (optionally transposed spatial)
__global__ void gate_pack_kernel(const float* __restrict__ a, const float* __restrict__ g,
                                 unsigned short* __restrict__ out, int trans) {
  const long long LLc = (long long)L_ * L_;
  const long long n = LLc * P_;
  for (long long i = blockIdx.x * 256LL + threadIdx.x; i < n; i += (long long)gridDim.x * 256) {
    const int d = (int)(i & (P_ - 1));
    const long long r = i >> 7;
    const long long x1 = r / L_, x2 = r - x1 * L_;
    const float v = a[i] * sigm(g[i]);
    const long long dst = trans ? ((long long)d * LLc + x2 * L_ + x1)
                                : ((long long)d * LLc + x1 * L_ + x2);
    out[dst] = f2bf(v);
  }
}
__global__ void attn_add_kernel(float* __restrict__ pair, const float* __restrict__ src, int trans) {
  const long long n = (long long)L_ * L_ * P_;
  for (long long i = blockIdx.x * 256LL + threadIdx.x; i < n; i += (long long)gridDim.x * 256) {
    const int d = (int)(i & (P_ - 1));
    const long long r = i >> 7;
    const long long ii = r / L_, jj = r - ii * L_;
    const long long sr = trans ? (jj * L_ + ii) : r;
    pair[i] += src[sr * P_ + d];
  }
}
__global__ void gate_add_kernel(float* __restrict__ pair, const float* __restrict__ o,
                                const float* __restrict__ gol) {
  const long long n = (long long)L_ * L_ * P_;
  for (long long i = blockIdx.x * 256LL + threadIdx.x; i < n; i += (long long)gridDim.x * 256)
    pair[i] += o[i] * sigm(gol[i]);
}
__global__ void h2_kernel(const float* __restrict__ U, const float* __restrict__ V,
                          unsigned short* __restrict__ out) {
  const long long n = (long long)L_ * L_ * P_;
  for (long long i = blockIdx.x * 256LL + threadIdx.x; i < n; i += (long long)gridDim.x * 256) {
    const int d = (int)(i & (P_ - 1));
    const long long r = i >> 7;
    const long long ii = r / L_, jj = r - ii * L_;
    out[i] = f2bf(geluf(U[ii * P_ + d] + V[jj * P_ + d]));
  }
}
__global__ __launch_bounds__(128)
void pool_w_kernel(const float* __restrict__ pn, const float* __restrict__ w,
                   const float* __restrict__ bptr, float* __restrict__ out) {
  const long long r = blockIdx.x;
  const int tid = threadIdx.x;
  __shared__ float red[128];
  red[tid] = pn[r * P_ + tid] * w[tid];
  __syncthreads();
  for (int o = 64; o > 0; o >>= 1) { if (tid < o) red[tid] += red[tid + o]; __syncthreads(); }
  if (tid == 0) out[r] = red[0] + bptr[0];
}
__global__ __launch_bounds__(128)
void pooled_kernel(const float* __restrict__ w, const float* __restrict__ pn,
                   float* __restrict__ out) {
  const int i = blockIdx.x, d = threadIdx.x;
  float acc = 0.f;
  for (int j = 0; j < L_; ++j)
    acc += w[i * L_ + j] * pn[((long long)i * L_ + j) * P_ + d];
  out[i * P_ + d] = acc;
}

// ---------------- host orchestration ----------------
static inline void* wsat(void* base, size_t off) { return (void*)((char*)base + off); }

extern "C" void kernel_launch(void* const* d_in, const int* in_sizes, int n_in,
                              void* d_out, int out_size, void* d_ws, size_t ws_size,
                              hipStream_t stream)
{
  (void)in_sizes; (void)out_size; (void)ws_size;
  const float* seq_in  = (const float*)d_in[0];
  const float* pair_in = (const float*)d_in[1];
  const float* PRM[96] = {};
  for (int i = 2; i < n_in && i < 98; ++i) PRM[i - 2] = (const float*)d_in[i];

  const long long LL  = (long long)L_ * L_;
  const long long LLP = LL * P_;

  size_t off = 0;
  auto alloc = [&](size_t bytes) {
    size_t o = off; off = (off + bytes + 255) & ~(size_t)255; return o;
  };

  // big scratch
  float* F1 = (float*)wsat(d_ws, alloc(LLP * 4));
  float* F2 = (float*)wsat(d_ws, alloc(LLP * 4));
  float* F3 = (float*)wsat(d_ws, alloc(LLP * 4));
  float* SC = (float*)wsat(d_ws, alloc(LLP * 4));      // scores chunk / 4th f32 temp
  unsigned short* B1 = (unsigned short*)wsat(d_ws, alloc(LLP * 2));
  unsigned short* B2 = (unsigned short*)wsat(d_ws, alloc(LLP * 2));
  unsigned short* B3 = (unsigned short*)wsat(d_ws, alloc(LLP * 2));
  unsigned short* QB = (unsigned short*)wsat(d_ws, alloc(LL * 512 * 2)); // qkv2 / pffn hidden
  // small scratch
  float* scS           = (float*)wsat(d_ws, alloc((long long)H_ * LL * 4));
  unsigned short* prb  = (unsigned short*)wsat(d_ws, alloc((long long)H_ * LL * 2));
  unsigned short* slb  = (unsigned short*)wsat(d_ws, alloc((long long)L_ * D_ * 2));
  unsigned short* qkvb = (unsigned short*)wsat(d_ws, alloc((long long)L_ * 3 * D_ * 2));
  float* oS            = (float*)wsat(d_ws, alloc((long long)L_ * D_ * 4));
  unsigned short* oSb  = (unsigned short*)wsat(d_ws, alloc((long long)L_ * D_ * 2));
  float* ffn1          = (float*)wsat(d_ws, alloc((long long)L_ * 2 * F_ * 4));
  unsigned short* xg   = (unsigned short*)wsat(d_ws, alloc((long long)L_ * F_ * 2));
  float* poolF         = (float*)wsat(d_ws, alloc((long long)L_ * P_ * 4));
  unsigned short* poolB= (unsigned short*)wsat(d_ws, alloc((long long)L_ * P_ * 2));
  unsigned short* snb  = (unsigned short*)wsat(d_ws, alloc((long long)L_ * D_ * 2));
  float* Uf            = (float*)wsat(d_ws, alloc((long long)L_ * P_ * 4));
  float* Vf            = (float*)wsat(d_ws, alloc((long long)L_ * P_ * 4));

  // bf16 weight copies
  unsigned short* WB[96] = {};
  auto cw = [&](int idx, long long cnt) {
    unsigned short* p = (unsigned short*)wsat(d_ws, alloc((size_t)cnt * 2));
    cvt_kernel<<<dim3(1024), dim3(256), 0, stream>>>(PRM[idx], p, cnt);
    WB[idx] = p;
  };
  cw(2, 768LL * 256);  cw(4, 256LL * 256);  cw(6, 8LL * 128);
  cw(10, 2048LL * 256); cw(12, 256LL * 1024);
  cw(16, 384LL * 128); cw(18, 128LL * 128);
  cw(22, 384LL * 128); cw(24, 128LL * 128);
  const int triW[12] = {30, 32, 34, 36, 38, 40, 46, 48, 50, 52, 54, 56};
  for (int i = 0; i < 12; ++i) cw(triW[i], 128LL * 128);
  cw(60, 512LL * 128); cw(62, 128LL * 512);
  cw(68, 256LL * 128); cw(73, 128LL * 512); cw(75, 128LL * 128);

  auto gemm = [&](const unsigned short* A, const unsigned short* B, const float* bias,
                  void* C, int cbf, int M, int N, int K,
                  long long aO, long long aI, int aR, int aK,
                  long long bO, long long bI, int bR, int bK,
                  long long cO, long long cI, int cR,
                  int batch, int bInner, float alpha, const float* gate, int accum, int act) {
    dim3 grid((N + 31) / 32, (M + 15) / 16, batch);
    if (aK == 1 && bK == 1)
      gemm_wmma_kernel<1, 1><<<grid, dim3(32), 0, stream>>>((const __bf16*)A, (const __bf16*)B,
          bias, C, cbf, M, N, K, aO, aI, aR, aK, bO, bI, bR, bK, cO, cI, cR,
          bInner, alpha, gate, accum, act);
    else if (aK == 1)
      gemm_wmma_kernel<1, 0><<<grid, dim3(32), 0, stream>>>((const __bf16*)A, (const __bf16*)B,
          bias, C, cbf, M, N, K, aO, aI, aR, aK, bO, bI, bR, bK, cO, cI, cR,
          bInner, alpha, gate, accum, act);
    else
      gemm_wmma_kernel<0, 0><<<grid, dim3(32), 0, stream>>>((const __bf16*)A, (const __bf16*)B,
          bias, C, cbf, M, N, K, aO, aI, aR, aK, bO, bI, bR, bK, cO, cI, cR,
          bInner, alpha, gate, accum, act);
  };
  auto gemmS = [&](const unsigned short* A, const unsigned short* B, const float* bias,
                   void* C, int cbf, int M, int N, int K, float alpha, int accum, int act,
                   const float* gate) {
    gemm(A, B, bias, C, cbf, M, N, K, 0, 0, K, 1, 0, 0, K, 1, 0, 0, N,
         1, 1, alpha, gate, accum, act);
  };

  float* outSeq  = (float*)d_out;
  float* outPair = outSeq + (long long)L_ * D_;
  hipMemcpyAsync(outSeq, seq_in, (size_t)L_ * D_ * 4, hipMemcpyDeviceToDevice, stream);
  hipMemcpyAsync(outPair, pair_in, (size_t)LLP * 4, hipMemcpyDeviceToDevice, stream);

  const float isq = 0.17677669529663688f;  // 1/sqrt(32)

  // ---- 1. seq self-attention with pair bias ----
  ln_kernel<<<L_, 128, 0, stream>>>(outSeq, PRM[0], PRM[1], nullptr, slb, D_, 0);
  cvt_kernel<<<2048, 256, 0, stream>>>(pair_in, B1, LLP);
  gemmS(B1, WB[6], PRM[7], SC, 0, (int)LL, H_, P_, 1.0f, 0, 0, nullptr);   // pbias (LL,8)
  gemmS(slb, WB[2], PRM[3], qkvb, 1, L_, 3 * D_, D_, 1.0f, 0, 0, nullptr); // qkv bf16
  gemm(qkvb, qkvb + D_, nullptr, scS, 0, L_, L_, 32,
       32, 0, 3 * D_, 1,
       32, 0, 3 * D_, 1,
       LL, 0, L_,
       H_, 1, isq, nullptr, 0, 0);
  softmax_kernel<<<H_ * L_, 128, 0, stream>>>(scS, L_, SC, L_, H_);
  cvt_kernel<<<2048, 256, 0, stream>>>(scS, prb, (long long)H_ * LL);
  gemm(prb, qkvb + 2 * D_, nullptr, oS, 0, L_, 32, L_,
       LL, 0, L_, 1,
       32, 0, 1, 3 * D_,
       32, 0, D_,
       H_, 1, 1.0f, nullptr, 0, 0);
  cvt_kernel<<<512, 256, 0, stream>>>(oS, oSb, (long long)L_ * D_);
  gemmS(oSb, WB[4], PRM[5], outSeq, 0, L_, D_, D_, 1.0f, 1, 0, nullptr);

  // ---- 2. seq GEGLU FFN ----
  ln_kernel<<<L_, 128, 0, stream>>>(outSeq, PRM[8], PRM[9], nullptr, slb, D_, 0);
  gemmS(slb, WB[10], PRM[11], ffn1, 0, L_, 2 * F_, D_, 1.0f, 0, 0, nullptr);
  geglu_kernel<<<1024, 256, 0, stream>>>(ffn1, xg, L_, F_);
  gemmS(xg, WB[12], PRM[13], outSeq, 0, L_, D_, F_, 1.0f, 1, 0, nullptr);

  // ---- 3/4. pair row + column attention ----
  auto pair_attn = [&](int ig, int iqkv, int io, int column) {
    ln_kernel<<<(int)LL, 128, 0, stream>>>(outPair, PRM[ig], PRM[ig + 1], nullptr, B1, P_,
                                           column ? L_ : 0);
    gemmS(B1, WB[iqkv], PRM[iqkv + 1], QB, 1, (int)LL, 3 * P_, P_, 1.0f, 0, 0, nullptr);
    const int CH = 32;
    for (int i0 = 0; i0 < L_; i0 += CH) {
      const unsigned short* qbase = QB + (long long)i0 * L_ * (3 * P_);
      gemm(qbase, qbase + P_, nullptr, SC, 0, L_, L_, 32,
           (long long)L_ * 3 * P_, 32, 3 * P_, 1,
           (long long)L_ * 3 * P_, 32, 3 * P_, 1,
           (long long)HP_ * LL, LL, L_,
           CH * HP_, HP_, isq, nullptr, 0, 0);
      softmax_kernel<<<CH * HP_ * L_, 128, 0, stream>>>(SC, L_, nullptr, 1, 1);
      cvt_kernel<<<2048, 256, 0, stream>>>(SC, B2, (long long)CH * HP_ * LL);
      gemm(B2, qbase + 2 * P_, nullptr, F1 + (long long)i0 * L_ * P_, 0, L_, 32, L_,
           (long long)HP_ * LL, LL, L_, 1,
           (long long)L_ * 3 * P_, 32, 1, 3 * P_,
           (long long)L_ * P_, 32, P_,
           CH * HP_, HP_, 1.0f, nullptr, 0, 0);
    }
    cvt_kernel<<<2048, 256, 0, stream>>>(F1, B1, LLP);
    gemmS(B1, WB[io], PRM[io + 1], F2, 0, (int)LL, P_, P_, 1.0f, 0, 0, nullptr);
    attn_add_kernel<<<2048, 256, 0, stream>>>(outPair, F2, column);
  };
  pair_attn(14, 16, 18, 0);
  pair_attn(20, 22, 24, 1);

  // ---- 5. triangular updates (128 channel-batched 384^3 WMMA GEMMs) ----
  auto tri = [&](int base, int outgoing) {
    const int n_g = base, no_g = base + 2, a_w = base + 4, b_w = base + 6,
              ga_w = base + 8, gb_w = base + 10, o_w = base + 12, go_w = base + 14;
    ln_kernel<<<(int)LL, 128, 0, stream>>>(outPair, PRM[n_g], PRM[n_g + 1], nullptr, B1, P_, 0);
    gemmS(B1, WB[go_w], PRM[go_w + 1], F2, 0, (int)LL, P_, P_, 1.0f, 0, 0, nullptr);
    gemmS(B1, WB[a_w], PRM[a_w + 1], F3, 0, (int)LL, P_, P_, 1.0f, 0, 0, nullptr);
    gemmS(B1, WB[ga_w], PRM[ga_w + 1], SC, 0, (int)LL, P_, P_, 1.0f, 0, 0, nullptr);
    gate_pack_kernel<<<2048, 256, 0, stream>>>(F3, SC, B2, 0);
    gemmS(B1, WB[b_w], PRM[b_w + 1], F3, 0, (int)LL, P_, P_, 1.0f, 0, 0, nullptr);
    gemmS(B1, WB[gb_w], PRM[gb_w + 1], SC, 0, (int)LL, P_, P_, 1.0f, 0, 0, nullptr);
    gate_pack_kernel<<<2048, 256, 0, stream>>>(F3, SC, B3, outgoing);
    gemm(B2, B3, nullptr, F3, 0, L_, L_, L_,
         LL, 0, L_, 1,
         LL, 0, L_, 1,
         LL, 0, L_,
         P_, 1, 1.0f, nullptr, 0, 0);
    ln_ch_kernel<<<(int)LL, 128, 0, stream>>>(F3, PRM[no_g], PRM[no_g + 1], B2, LL);
    gemmS(B2, WB[o_w], PRM[o_w + 1], F1, 0, (int)LL, P_, P_, 1.0f, 0, 0, nullptr);
    gate_add_kernel<<<2048, 256, 0, stream>>>(outPair, F1, F2);
  };
  tri(26, 1);  // outgoing
  tri(42, 0);  // incoming

  // ---- 6. pair FFN ----
  ln_kernel<<<(int)LL, 128, 0, stream>>>(outPair, PRM[58], PRM[59], nullptr, B1, P_, 0);
  gemmS(B1, WB[60], PRM[61], QB, 1, (int)LL, 4 * P_, P_, 1.0f, 0, 1, nullptr); // gelu epi
  gemmS(QB, WB[62], PRM[63], outPair, 0, (int)LL, P_, 4 * P_, 1.0f, 1, 0, nullptr);

  // ---- 7. seq<->pair cross update ----
  ln_kernel<<<(int)LL, 128, 0, stream>>>(outPair, PRM[64], PRM[65], F1, nullptr, P_, 0);
  pool_w_kernel<<<(int)LL, 128, 0, stream>>>(F1, PRM[66], PRM[67], SC);
  softmax_kernel<<<L_, 128, 0, stream>>>(SC, L_, nullptr, 1, 1);
  pooled_kernel<<<L_, 128, 0, stream>>>(SC, F1, poolF);
  cvt_kernel<<<192, 256, 0, stream>>>(poolF, poolB, (long long)L_ * P_);
  gemmS(poolB, WB[68], PRM[69], outSeq, 0, L_, D_, P_, 1.0f, 1, 0, PRM[70]);
  ln_kernel<<<L_, 128, 0, stream>>>(outSeq, PRM[71], PRM[72], nullptr, snb, D_, 0);
  gemm(snb, WB[73], PRM[74], Uf, 0, L_, P_, D_,
       0, 0, D_, 1, 0, 0, 2 * D_, 1, 0, 0, P_, 1, 1, 1.0f, nullptr, 0, 0);
  gemm(snb, WB[73] + D_, nullptr, Vf, 0, L_, P_, D_,
       0, 0, D_, 1, 0, 0, 2 * D_, 1, 0, 0, P_, 1, 1, 1.0f, nullptr, 0, 0);
  h2_kernel<<<2048, 256, 0, stream>>>(Uf, Vf, B1);
  gemmS(B1, WB[75], PRM[76], outPair, 0, (int)LL, P_, P_, 1.0f, 1, 0, PRM[77]);
}
